// MPSMemoryEncoder_5033701671197
// MI455X (gfx1250) — matrix operand order repaired and checked by reading, not compiled, not run on hardware
//
#include <hip/hip_runtime.h>
#include <hip/hip_fp16.h>
#include <stdint.h>

typedef __attribute__((ext_vector_type(16))) _Float16 v16h;
typedef __attribute__((ext_vector_type(8)))  float    v8f;
typedef __attribute__((ext_vector_type(4)))  float    vf4;
typedef __attribute__((ext_vector_type(4)))  uint32_t u32x4;
typedef __attribute__((ext_vector_type(4)))  int32_t  i32x4;
typedef __attribute__((ext_vector_type(8)))  int32_t  i32x8;

#define N_BATCH 2048
#define N_SITES 128
#define N_FEAT  64
#define N_D     32
#define N_OUT   256

// Workspace layout (f16):
//  region A (core WMMA-B fragments): [site][d][fh][rt][lane][16 halves]  = 128 * 131072 B = 16 MB
//  region X (x WMMA-A fragments):    [btile][site][fh][lane][16 halves]  = 128 * 128 * 2048 B = 32 MB
#define WS_A_BYTES (N_SITES * 131072)
#define WS_X_OFF   WS_A_BYTES

// ---------------- pre-pass 1: cores f32 -> f16 in B-fragment layout ----------------
// B (K=32 x N=16) f16 fragment layout: lane L holds column r = rt*16 + (L&15),
// halves j=0..15 hold K = (L>=16 ? 16 : 0) + j  (K == f offset within the 32-window).
__global__ __launch_bounds__(256) void prep_cores(const float* __restrict__ cores,
                                                  _Float16* __restrict__ dst) {
    uint32_t h    = blockIdx.x * 256u + threadIdx.x;   // one output half per thread
    uint32_t j    = h & 15u;
    uint32_t lane = (h >> 4) & 31u;
    uint32_t rt   = (h >> 9) & 1u;
    uint32_t fh   = (h >> 10) & 1u;
    uint32_t d    = (h >> 11) & 31u;
    uint32_t site = h >> 16;
    uint32_t f    = fh * 32u + ((lane >> 4) << 4) + j;
    uint32_t r    = rt * 16u + (lane & 15u);
    dst[h] = (_Float16)cores[(((site * 32u + d) * 64u) + f) * 32u + r];
}

// ---------------- pre-pass 2: x f32 -> f16 in A-fragment layout ----------------
// A (M=16 x K=32) f16 layout: lane L row m=L&15; halves hold K offsets
// {0..15} for L<16 and {16..31} for L>=16 (matches B fragment K order).
__global__ __launch_bounds__(256) void prep_x(const float* __restrict__ x,
                                              _Float16* __restrict__ dst) {
    uint32_t u     = blockIdx.x * 256u + threadIdx.x;
    uint32_t lane  = u & 31u;
    uint32_t fh    = (u >> 5) & 1u;
    uint32_t site  = (u >> 6) & 127u;
    uint32_t btile = u >> 13;
    uint32_t m     = lane & 15u;
    uint32_t row   = btile * 16u + m;
    uint32_t fs    = fh * 32u + ((lane >> 4) << 4);
    const float* src = x + ((size_t)(row * 128u + site)) * 64u;
    v16h o;
#pragma unroll
    for (int j = 0; j < 8; ++j) o[j]     = (_Float16)src[fs + j];
#pragma unroll
    for (int j = 0; j < 8; ++j) o[8 + j] = (_Float16)src[fs + 8 + j];
    *(v16h*)(dst + (size_t)u * 16u) = o;
}

// ---------------- gfx1250 async global->LDS (ASYNCcnt path) ----------------
__device__ __forceinline__ void async_b128(uint32_t lds_off, const void* gaddr) {
    asm volatile("global_load_async_to_lds_b128 %0, %1, off"
                 :: "v"(lds_off), "v"(gaddr) : "memory");
}
__device__ __forceinline__ void wait_async0() {
    asm volatile("s_wait_asynccnt 0" ::: "memory");
}

// ---------------- gfx1250 Tensor Data Mover (TENSORcnt path) ----------------
// One descriptor copies a contiguous 128 KB tile (16384 x 8-byte elements) to LDS.
__device__ __forceinline__ void tdm_copy_128k(const void* gsrc, uint32_t lds_off) {
    uint64_t ga = (uint64_t)(uintptr_t)gsrc;
    u32x4 g0;
    g0[0] = 1u;                                   // count=1 (valid user descriptor)
    g0[1] = lds_off;                              // lds_addr (bytes)
    g0[2] = (uint32_t)ga;                         // global_addr[31:0]
    g0[3] = (uint32_t)(ga >> 32) | (2u << 30);    // global_addr[56:32] | type=2 ("image")
    i32x8 g1;
    g1[0] = (int)(3u << 16);                      // data_size=3 (8 B); no multicast/pad/iterate
    g1[1] = (int)(16384u << 16);                  // tensor_dim0[15:0] -> bits 63:48
    g1[2] = (int)(1u << 16);                      // tensor_dim0[31:16]=0 ; tensor_dim1[15:0]=1
    g1[3] = (int)(16384u << 16);                  // tensor_dim1[31:16]=0 ; tile_dim0=16384
    g1[4] = 1;                                    // tile_dim1=1 ; tile_dim2=0
    g1[5] = 16384;                                // tensor_dim0_stride[31:0]
    g1[6] = 0;                                    // stride hi / dim1_stride lo
    g1[7] = 0;
    i32x4 gz = {0, 0, 0, 0};
#if __clang_major__ >= 23
    i32x8 gz8 = {0, 0, 0, 0, 0, 0, 0, 0};
    __builtin_amdgcn_tensor_load_to_lds(g0, g1, gz, gz, gz8, 0);
#else
    __builtin_amdgcn_tensor_load_to_lds(g0, g1, gz, gz, 0);
#endif
}

// ---------------- fused MPS scan kernel ----------------
// grid 64, block 256 (8 waves): wave w -> tile tg=w>>2 (16 batch rows), K-group kg=w&3.
// Dynamic LDS map (bytes):
#define L_CORE0 0u
#define L_CORE1 131072u
#define L_XB0   262144u
#define L_XB1   266240u
#define L_RES0  270336u
#define L_RES1  274432u
#define L_VALS  278528u
#define L_TOTAL 278656u

__global__ __launch_bounds__(256, 1)
void mps_fused(const _Float16* __restrict__ wsA, const _Float16* __restrict__ wsX,
               const float* __restrict__ startv, const float* __restrict__ endv,
               const float* __restrict__ fcw, const float* __restrict__ fcb,
               float* __restrict__ out) {
    extern __shared__ char smem[];
    const uint32_t t    = threadIdx.x;
    const uint32_t lane = t & 31u;
    const uint32_t w    = t >> 5;
    const uint32_t kg   = w & 3u;
    const uint32_t tg   = w >> 2;
    const uint32_t smembase = (uint32_t)(uintptr_t)smem;   // LDS byte offset (low 32 bits)

    // init res ping buffer 0 with broadcast 'start' row
    {
        float* r0 = (float*)(smem + L_RES0);
#pragma unroll
        for (int i = 0; i < 8; ++i) {
            uint32_t idx = t * 8u + i;
            r0[idx] = startv[idx & 31u];
        }
    }

    // x fragments for one site: 2 tiles x 2048 B -> one b128 async per thread
    auto prefetch_x = [&](int site, uint32_t buf) {
        uint32_t btile = blockIdx.x * 2u + (t >> 7);
        const char* srcX = (const char*)wsX +
            ((size_t)(btile * 128u + (uint32_t)site)) * 2048u + (t & 127u) * 16u;
        async_b128(smembase + (buf ? L_XB1 : L_XB0) + t * 16u, srcX);
    };
    // core fragments for one site: single 128 KB TDM descriptor, issued by wave 0
    auto prefetch_core = [&](int site, uint32_t buf) {
        if (w == 0) {
            tdm_copy_128k((const char*)wsA + (size_t)site * 131072u,
                          smembase + (buf ? L_CORE1 : L_CORE0));
        }
    };

    prefetch_core(0, 0);
    prefetch_x(0, 0);
    if (w == 0) __builtin_amdgcn_s_wait_tensorcnt((short)0);
    wait_async0();
    __syncthreads();

    uint32_t respar = 0;
    for (int site = 0; site < N_SITES; ++site) {
        const uint32_t buf = (uint32_t)site & 1u;

        // previous res (f32 in LDS) -> 8 f16 scalars for this wave's d-slice
        const float* resOld = (const float*)(smem + (respar ? L_RES1 : L_RES0)) +
                              (tg * 16u + (lane & 15u)) * 32u;
        vf4 ra = *(const vf4*)(resOld + kg * 8u);
        vf4 rb = *(const vf4*)(resOld + kg * 8u + 4u);
        _Float16 rh[8];
#pragma unroll
        for (int i = 0; i < 4; ++i) { rh[i] = (_Float16)ra[i]; rh[4 + i] = (_Float16)rb[i]; }

        // x A-fragments for this tile
        const char* xb = smem + (buf ? L_XB1 : L_XB0);
        v16h xf0 = *(const v16h*)(xb + ((tg * 2u + 0u) * 32u + lane) * 32u);
        v16h xf1 = *(const v16h*)(xb + ((tg * 2u + 1u) * 32u + lane) * 32u);

        // zero next res buffer (one wave per tile)
        float* resNext = (float*)(smem + (respar ? L_RES0 : L_RES1));
        if (kg == 0) {
            vf4 z = {0.f, 0.f, 0.f, 0.f};
#pragma unroll
            for (int i = 0; i < 4; ++i)
                *((vf4*)(resNext + tg * 512u) + lane * 4u + i) = z;
        }

        if (site + 1 < N_SITES) {
            prefetch_core(site + 1, buf ^ 1u);
            prefetch_x(site + 1, buf ^ 1u);
        }

        // ---- 16 K-steps x 2 N-tiles of v_wmma_f32_16x16x32_f16 ----
        // B fragments register-prefetched one K-step ahead to give ds waits slack.
        v8f acc0 = {}; v8f acc1 = {};
        const char* cb = smem + (buf ? L_CORE1 : L_CORE0) + kg * 8u * 4096u + lane * 32u;
        v16h b0 = *(const v16h*)(cb);
        v16h b1 = *(const v16h*)(cb + 1024);
#pragma unroll
        for (int ts = 0; ts < 16; ++ts) {
            const int dl = ts >> 1, fh = ts & 1;
            v16h nb0, nb1;
            if (ts < 15) {
                const int ndl = (ts + 1) >> 1, nfh = (ts + 1) & 1;
                const char* np = cb + ndl * 4096 + nfh * 2048;
                nb0 = *(const v16h*)(np);
                nb1 = *(const v16h*)(np + 1024);
            }
            v16h a = (fh ? xf1 : xf0) * rh[dl];          // A = res[m,d] * x[m,f]
            acc0 = __builtin_amdgcn_wmma_f32_16x16x32_f16(false, a, false, b0,
                                                          (short)0, acc0, false, false);
            acc1 = __builtin_amdgcn_wmma_f32_16x16x32_f16(false, a, false, b1,
                                                          (short)0, acc1, false, false);
            b0 = nb0; b1 = nb1;
        }

        __syncthreads();   // zeros visible before accumulation

        // reduce K-split partials: C layout -> (m = i + 8*(lane>=16), n = lane&15)
        float* dstv = resNext + tg * 512u;
        const uint32_t mhi = (lane >> 4) * 8u;
        const uint32_t n   = lane & 15u;
#pragma unroll
        for (int i = 0; i < 8; ++i) {
            atomicAdd(dstv + (mhi + i) * 32u + n,        acc0[i]);
            atomicAdd(dstv + (mhi + i) * 32u + 16u + n,  acc1[i]);
        }

        if (w == 0) __builtin_amdgcn_s_wait_tensorcnt((short)0);  // core tile landed
        wait_async0();                                            // x tile landed
        __syncthreads();                                          // publish to workgroup
        respar ^= 1u;
    }

    // ---- epilogue: val = res . end ; out = val * fc_w + fc_b ----
    float* vals = (float*)(smem + L_VALS);
    if (kg == 0 && lane < 16u) {
        const float* rr = (const float*)(smem + L_RES0) + (tg * 16u + lane) * 32u;
        float acc = 0.f;
#pragma unroll
        for (int d = 0; d < 32; ++d) acc += rr[d] * endv[d];
        vals[tg * 16u + lane] = acc;
    }
    __syncthreads();
    {
        uint32_t rowl = t >> 3;
        uint32_t o0   = (t & 7u) * 32u;
        float val = vals[rowl];
        float* op = out + ((size_t)(blockIdx.x * 32u + rowl)) * 256u + o0;
#pragma unroll
        for (int i = 0; i < 8; ++i) {
            vf4 w4 = *(const vf4*)(fcw + o0 + i * 4u);
            vf4 b4 = *(const vf4*)(fcb + o0 + i * 4u);
            vf4 o4 = val * w4 + b4;
            *(vf4*)(op + i * 4u) = o4;
        }
    }
}

// ---------------- host launcher ----------------
extern "C" void kernel_launch(void* const* d_in, const int* in_sizes, int n_in,
                              void* d_out, int out_size, void* d_ws, size_t ws_size,
                              hipStream_t stream) {
    (void)in_sizes; (void)n_in; (void)out_size; (void)ws_size;
    const float* x      = (const float*)d_in[0];
    const float* cores  = (const float*)d_in[1];
    const float* startv = (const float*)d_in[2];
    const float* endv   = (const float*)d_in[3];
    const float* fcw    = (const float*)d_in[4];
    const float* fcb    = (const float*)d_in[5];
    float*       out    = (float*)d_out;

    _Float16* wsA = (_Float16*)d_ws;                              // 16 MB
    _Float16* wsX = (_Float16*)((char*)d_ws + WS_X_OFF);          // 32 MB  (needs 48 MB ws)

    prep_cores<<<(N_SITES * 131072 / 2) / 256, 256, 0, stream>>>(cores, wsA);
    prep_x<<<(N_BATCH / 16) * N_SITES * 2 * 32 / 256, 256, 0, stream>>>(x, wsX);

    mps_fused<<<N_BATCH / 32, 256, L_TOTAL, stream>>>(wsA, wsX, startv, endv, fcw, fcb, out);
}
